// NewtonNet_33535104648020
// MI455X (gfx1250) — matrix operand order repaired and checked by reading, not compiled, not run on hardware
//
#include <hip/hip_runtime.h>
#include <hip/hip_bf16.h>
#include <math.h>

// ---------------- problem constants ----------------
#define BB   16
#define AA   60
#define NNE  59
#define FF   128
#define RESN 20
#define KRBF 32                  // RES padded to 32 for one WMMA k-step
#define NODE (BB*AA)             // 960   (= 15 * 64)
#define EDGE (BB*AA*NNE)         // 56640 (= 885 * 64)
#define CUTF 5.0f
#define PIF  3.14159265358979f

typedef __attribute__((ext_vector_type(16))) _Float16 v16h;
typedef __attribute__((ext_vector_type(8)))  float    v8f;
typedef __attribute__((ext_vector_type(4)))  int      v4i;

#if __has_builtin(__builtin_amdgcn_global_load_async_to_lds_b128) && \
    __has_builtin(__builtin_amdgcn_s_wait_asynccnt)
#define USE_ASYNC_LDS 1
#define AS1 __attribute__((address_space(1)))
#define AS3 __attribute__((address_space(3)))
#endif

// =====================================================================
// weight pre-pass: f32 [K,N] row-major  ->  f16 transposed [N][Kpad],
// zero-padded in K.  Done once per call; makes GEMM staging a raw copy.
// =====================================================================
__global__ void k_cvtw(const float* __restrict__ W, _Float16* __restrict__ Wt,
                       int Kpad, int Kw, int N) {
    int idx = blockIdx.x * blockDim.x + threadIdx.x;
    if (idx >= N * Kpad) return;
    int n = idx / Kpad, k = idx - n * Kpad;
    Wt[idx] = (_Float16)((k < Kw) ? W[k * N + n] : 0.0f);
}

// =====================================================================
// WMMA GEMM:  C[M,N] = epilogue(A[M,K] @ W[K,N] + bias)
//   - wave32, v_wmma_f32_16x16x32_f16, f32 accumulate
//   - block = 128 threads = 4 waves; each wave: 16 rows x (NT*16) cols
//   - Wt is pre-transposed f16 [N][K]; staged to LDS once per block via
//     GLOBAL_LOAD_ASYNC_TO_LDS_B128 (ASYNCcnt) when available
//   - grid-strides over 64-row strips to amortize the weight stage
//   - epilogue: +bias (opt), SiLU (opt), * rowscale[row] (opt)
// =====================================================================
template<int NT>
__global__ __launch_bounds__(128)
void gemm_wmma(const float* __restrict__ A, int lda,
               const _Float16* __restrict__ Wt, int K,
               const float* __restrict__ bias,
               const float* __restrict__ rowscale,
               float* __restrict__ C, int ldc, int act, int strips) {
    constexpr int N = NT * 16;
    __shared__ __align__(32) _Float16 sW[128 * 128];

    // ---- stage Wt (f16 [N][K]) -> LDS ----
    const int bytes = N * K * 2;
#ifdef USE_ASYNC_LDS
    {
        const char* g = (const char*)Wt;
        char*       l = (char*)sW;
        for (int off = threadIdx.x * 16; off < bytes; off += 128 * 16) {
            __builtin_amdgcn_global_load_async_to_lds_b128(
                (AS1 v4i*)(g + off), (AS3 v4i*)(l + off), 0, 0);
        }
        __builtin_amdgcn_s_wait_asynccnt(0);
    }
#else
    {
        const uint4* src = (const uint4*)Wt;
        uint4*       dst = (uint4*)sW;
        for (int i = threadIdx.x; i < bytes / 16; i += 128) dst[i] = src[i];
    }
#endif
    __syncthreads();

    const int wid  = threadIdx.x >> 5;
    const int lane = threadIdx.x & 31;
    const int hl   = lane >> 4;       // half-wave select
    const int l15  = lane & 15;

    for (int strip = blockIdx.x; strip < strips; strip += gridDim.x) {
        const long tile_m = (long)strip * 64 + wid * 16;
        const float* Arow = A + (tile_m + l15) * (long)lda;

        // prefetch next strip's A rows (global_prefetch_b8)
        if (strip + gridDim.x < strips) {
            const float* nxt = A + (((long)(strip + gridDim.x) * 64 + wid * 16 + l15)) * (long)lda;
            __builtin_prefetch(nxt, 0, 0);
        }

        v8f acc[NT] = {};

        for (int k0 = 0; k0 < K; k0 += 32) {
            // A fragment: contiguous k = 16*hl + s (16x32 per wave)
            const int r0 = k0 + hl * 16;
            const int r1 = r0 + 8;
            float4 x0 = *(const float4*)(Arow + r0);
            float4 x1 = *(const float4*)(Arow + r0 + 4);
            float4 y0 = *(const float4*)(Arow + r1);
            float4 y1 = *(const float4*)(Arow + r1 + 4);
            v16h af;
            af[0]  = (_Float16)x0.x; af[1]  = (_Float16)x0.y;
            af[2]  = (_Float16)x0.z; af[3]  = (_Float16)x0.w;
            af[4]  = (_Float16)x1.x; af[5]  = (_Float16)x1.y;
            af[6]  = (_Float16)x1.z; af[7]  = (_Float16)x1.w;
            af[8]  = (_Float16)y0.x; af[9]  = (_Float16)y0.y;
            af[10] = (_Float16)y0.z; af[11] = (_Float16)y0.w;
            af[12] = (_Float16)y1.x; af[13] = (_Float16)y1.y;
            af[14] = (_Float16)y1.z; af[15] = (_Float16)y1.w;

            const int kb = k0 + hl * 16;
#pragma unroll
            for (int t = 0; t < NT; ++t) {
                const int col = t * 16 + l15;
                v16h bf = *(const v16h*)(sW + col * K + kb);
                acc[t] = __builtin_amdgcn_wmma_f32_16x16x32_f16(
                    false, af, false, bf, (short)0, acc[t], false, false);
            }
        }

        // ---- epilogue + store (row = tile + hl*8 + r, col = t*16+l15) ----
        const long rbase = tile_m + hl * 8;
#pragma unroll
        for (int t = 0; t < NT; ++t) {
            const int col = t * 16 + l15;
            const float bcol = bias ? bias[col] : 0.0f;
#pragma unroll
            for (int r = 0; r < 8; ++r) {
                long row = rbase + r;
                float x = acc[t][r] + bcol;
                if (act) x = x / (1.0f + __expf(-x));          // SiLU
                if (rowscale) x *= rowscale[row];
                C[row * (long)ldc + col] = x;
            }
        }
    }
}

// =====================================================================
// elementwise / reduction kernels
// =====================================================================
__global__ void k_zero(float* p, int n) {
    int i = blockIdx.x * blockDim.x + threadIdx.x;
    if (i < n) p[i] = 0.0f;
}

__global__ void k_embed(const int* __restrict__ Z, const float* __restrict__ emb,
                        float* __restrict__ a) {
    int idx = blockIdx.x * blockDim.x + threadIdx.x;   // NODE*FF
    int node = idx >> 7, f = idx & 127;
    a[idx] = emb[Z[node] * FF + f];
}

// bessel rbf (padded to 32) + polynomial cutoff
__global__ void k_rbf(const float* __restrict__ D, float* __restrict__ rbfpad,
                      float* __restrict__ cut) {
    int idx = blockIdx.x * blockDim.x + threadIdx.x;   // EDGE*32
    if (idx >= EDGE * KRBF) return;
    int e = idx >> 5, n = idx & 31;
    float d  = D[e];
    float dm = fmaxf(d, 1e-8f);
    float val = 0.0f;
    if (n < RESN) {
        val = 0.632455532f * __sinf((float)(n + 1) * PIF * dm / CUTF) / dm;
    }
    rbfpad[idx] = val;
    if (n == 0) {
        float x  = d / CUTF;
        float x2 = x * x, x4 = x2 * x2, x8 = x4 * x4;
        float x9 = x8 * x, x10 = x9 * x, x11 = x10 * x;
        float o = 1.0f - 55.0f * x9 + 99.0f * x10 - 45.0f * x11;
        cut[e] = (x < 1.0f) ? o : 0.0f;
    }
}

// msij = rbf_msij * gather(a_m) * a_m ; a += masked sum over neighbors
__global__ void k_msij(float* __restrict__ E1, const float* __restrict__ a_m,
                       const int* __restrict__ Nbr, const float* __restrict__ NM,
                       float* __restrict__ a) {
    int node = blockIdx.x;            // b*AA + i
    int f    = threadIdx.x;           // 0..127
    int b    = node / AA;
    float am_i = a_m[node * FF + f];
    float s = 0.0f;
    const int*   nrow = Nbr + node * NNE;
    const float* mrow = NM  + node * NNE;
    for (int j = 0; j < NNE; ++j) {
        long e = (long)node * NNE + j;
        float m = E1[e * FF + f] * a_m[(b * AA + nrow[j]) * FF + f] * am_i;
        E1[e * FF + f] = m;
        s += m * mrow[j];
    }
    a[node * FF + f] += s;
}

// F_ij[e,x] = dot(msij[e,:], wf) * V[e,x]
__global__ void k_phif(const float* __restrict__ msij, const float* __restrict__ wf,
                       const float* __restrict__ V, float* __restrict__ Fij) {
    int node = blockIdx.x;
    int wid  = threadIdx.x >> 5;
    int lane = threadIdx.x & 31;
    for (int j = wid; j < NNE; j += 4) {
        long e = (long)node * NNE + j;
        float s = 0.0f;
        for (int c = lane; c < FF; c += 32) s += msij[e * FF + c] * wf[c];
#pragma unroll
        for (int off = 16; off > 0; off >>= 1) s += __shfl_xor(s, off, 32);
        if (lane < 3) Fij[e * 3 + lane] = s * V[e * 3 + lane];
    }
}

// F_i[node,x,f] = sum_j NM * fscale[e,f] * Fij[e,x]
__global__ void k_Fi(const float* __restrict__ FS, const float* __restrict__ Fij,
                     const float* __restrict__ NM, float* __restrict__ F_i) {
    int node = blockIdx.x, f = threadIdx.x;
    float a0 = 0.f, a1 = 0.f, a2 = 0.f;
    for (int j = 0; j < NNE; ++j) {
        long e = (long)node * NNE + j;
        float c = NM[e] * FS[e * FF + f];
        a0 += c * Fij[e * 3 + 0];
        a1 += c * Fij[e * 3 + 1];
        a2 += c * Fij[e * 3 + 2];
    }
    F_i[((long)node * 3 + 0) * FF + f] = a0;
    F_i[((long)node * 3 + 1) * FF + f] = a1;
    F_i[((long)node * 3 + 2) * FF + f] = a2;
}

// dr_ext[node,x,f] = sum_j NM * rext[e,f] * r_dyn[nbr,x,f]
__global__ void k_drext(const float* __restrict__ RX, const int* __restrict__ Nbr,
                        const float* __restrict__ NM, const float* __restrict__ r_dyn,
                        float* __restrict__ drext) {
    int node = blockIdx.x, f = threadIdx.x;
    int b = node / AA;
    float a0 = 0.f, a1 = 0.f, a2 = 0.f;
    for (int j = 0; j < NNE; ++j) {
        long e = (long)node * NNE + j;
        float c = NM[e] * RX[e * FF + f];
        long base = ((long)(b * AA + Nbr[e]) * 3) * FF + f;
        a0 += c * r_dyn[base];
        a1 += c * r_dyn[base + FF];
        a2 += c * r_dyn[base + 2 * FF];
    }
    drext[((long)node * 3 + 0) * FF + f] = a0;
    drext[((long)node * 3 + 1) * FF + f] = a1;
    drext[((long)node * 3 + 2) * FF + f] = a2;
}

// f_dyn += F_i ; r_dyn += phi_r_out * F_i + dr_ext
__global__ void k_update_dyn(const float* __restrict__ F_i, const float* __restrict__ drext,
                             const float* __restrict__ pr, float* __restrict__ f_dyn,
                             float* __restrict__ r_dyn) {
    int node = blockIdx.x, f = threadIdx.x;
    float p = pr[node * FF + f];
#pragma unroll
    for (int x = 0; x < 3; ++x) {
        long idx = ((long)node * 3 + x) * FF + f;
        float fi = F_i[idx];
        f_dyn[idx] += fi;
        r_dyn[idx] += p * fi + drext[idx];
    }
}

// a += phi_e_out * (-sum_x f_dyn*r_dyn)
__global__ void k_energy(const float* __restrict__ pe, const float* __restrict__ f_dyn,
                         const float* __restrict__ r_dyn, float* __restrict__ a) {
    int node = blockIdx.x, f = threadIdx.x;
    float s = 0.f;
#pragma unroll
    for (int x = 0; x < 3; ++x) {
        long idx = ((long)node * 3 + x) * FF + f;
        s += f_dyn[idx] * r_dyn[idx];
    }
    a[node * FF + f] += pe[node * FF + f] * (-s);
}

// out[b] = sum_i (h2[b,i,:] . w2 + b2) * AM[b,i]
__global__ void k_final(const float* __restrict__ h2, const float* __restrict__ w2,
                        const float* __restrict__ b2, const float* __restrict__ AM,
                        float* __restrict__ out) {
    __shared__ float red[64];
    int b = blockIdx.x, t = threadIdx.x;
    float accB = 0.f;
    for (int i = 0; i < AA; ++i) {
        int node = b * AA + i;
        red[t] = h2[(long)node * 64 + t] * w2[t];
        __syncthreads();
        for (int s = 32; s > 0; s >>= 1) {
            if (t < s) red[t] += red[t + s];
            __syncthreads();
        }
        if (t == 0) accB += (red[0] + b2[0]) * AM[node];
        __syncthreads();
    }
    if (t == 0) out[b] = accB;
}

// =====================================================================
extern "C" void kernel_launch(void* const* d_in, const int* in_sizes, int n_in,
                              void* d_out, int out_size, void* d_ws, size_t ws_size,
                              hipStream_t stream) {
    (void)in_sizes; (void)n_in; (void)out_size; (void)ws_size;
    const int*   Z   = (const int*)d_in[0];
    const int*   Nbr = (const int*)d_in[2];
    const float* NM  = (const float*)d_in[3];
    const float* AM  = (const float*)d_in[4];
    const float* D   = (const float*)d_in[5];
    const float* V   = (const float*)d_in[6];
    auto P = [&](int i) { return (const float*)d_in[i]; };
    // params pytree, dict keys alphabetical: ae, embedding, layers
    const float *bae0 = P(7),  *Wae0 = P(8),  *bae1 = P(9), *Wae1 = P(10);
    const float *bae2 = P(11), *Wae2 = P(12), *emb  = P(13);

    float* ws = (float*)d_ws;
    float* a_buf  = ws;                       // NODE*FF
    float* a_m    = a_buf  + NODE * FF;
    float* ntmp   = a_m    + NODE * FF;
    float* pr     = ntmp   + NODE * FF;
    float* pe     = pr     + NODE * FF;
    float* F_i    = pe     + NODE * FF;       // NODE*3*FF
    float* drext  = F_i    + NODE * 3 * FF;
    float* f_dyn  = drext  + NODE * 3 * FF;
    float* r_dyn  = f_dyn  + NODE * 3 * FF;
    float* h2     = r_dyn  + NODE * 3 * FF;   // NODE*64
    float* cut    = h2     + NODE * 64;       // EDGE
    float* Fij    = cut    + EDGE;            // EDGE*3
    float* rbfpad = Fij    + EDGE * 3;        // EDGE*KRBF
    float* E1     = rbfpad + (long)EDGE * KRBF;  // EDGE*FF
    float* E2     = E1     + (long)EDGE * FF;    // EDGE*FF
    // f16 weight pool: 35 slots of 128*128 halves
    _Float16* wh  = (_Float16*)(E2 + (long)EDGE * FF);
    auto WT = [&](int slot) { return wh + (long)slot * 128 * 128; };

    const int SE = EDGE / 64;   // 885 strips for edge GEMMs
    const int SN = NODE / 64;   // 15 strips for node GEMMs
    const int GE = 240;         // edge GEMM grid (grid-stride, ~4 strips/block)
    const int GN = SN;

    // ---- weight pre-pass: convert + transpose + pad to f16 ----
    auto cvtw = [&](const float* src, _Float16* dst, int Kpad, int Kw, int N) {
        k_cvtw<<<(N * Kpad + 255) / 256, 256, 0, stream>>>(src, dst, Kpad, Kw, N);
    };
    for (int l = 0; l < 3; ++l) {
        const int o = 14 + 21 * l, s = l * 11;
        cvtw(P(o+20), WT(s+0), KRBF, RESN, FF);  // Wrbf
        cvtw(P(o+1),  WT(s+1), FF, FF, FF);      // Wa0
        cvtw(P(o+3),  WT(s+2), FF, FF, FF);      // Wa1
        cvtw(P(o+10), WT(s+3), FF, FF, FF);      // Wfs0
        cvtw(P(o+12), WT(s+4), FF, FF, FF);      // Wfs1
        cvtw(P(o+14), WT(s+5), FF, FF, FF);      // Wr0
        cvtw(P(o+16), WT(s+6), FF, FF, FF);      // Wr1
        cvtw(P(o+17), WT(s+7), FF, FF, FF);      // Wrx0
        cvtw(P(o+18), WT(s+8), FF, FF, FF);      // Wrx1
        cvtw(P(o+5),  WT(s+9), FF, FF, FF);      // We0
        cvtw(P(o+7),  WT(s+10), FF, FF, FF);     // We1
    }
    cvtw(Wae0, WT(33), FF, FF, FF);
    cvtw(Wae1, WT(34), FF, FF, 64);

    // init
    k_zero<<<(2 * NODE * 3 * FF + 255) / 256, 256, 0, stream>>>(f_dyn, 2 * NODE * 3 * FF);
    k_embed<<<NODE, 128, 0, stream>>>(Z, emb, a_buf);
    k_rbf<<<(EDGE * KRBF + 127) / 128, 128, 0, stream>>>(D, rbfpad, cut);

    for (int l = 0; l < 3; ++l) {
        const int o = 14 + 21 * l, s = l * 11;
        const float *ba0 = P(o+0),  *ba1 = P(o+2);
        const float *be0 = P(o+4),  *be1 = P(o+6);
        const float *Wf  = P(o+8);
        const float *bfs0= P(o+9),  *bfs1= P(o+11);
        const float *br0 = P(o+13), *br1 = P(o+15);
        const float *brbf= P(o+19);

        // rbf_msij = (rbf @ Wrbf + b) * cutoff
        gemm_wmma<8><<<GE, 128, 0, stream>>>(rbfpad, KRBF, WT(s+0), KRBF, brbf, cut, E1, FF, 0, SE);
        // a_m = mlp2(a)
        gemm_wmma<8><<<GN, 128, 0, stream>>>(a_buf, FF, WT(s+1), FF, ba0, nullptr, ntmp, FF, 1, SN);
        gemm_wmma<8><<<GN, 128, 0, stream>>>(ntmp,  FF, WT(s+2), FF, ba1, nullptr, a_m,  FF, 0, SN);
        // msij + a update
        k_msij<<<NODE, 128, 0, stream>>>(E1, a_m, Nbr, NM, a_buf);
        // F_ij = (msij @ wf) * V
        k_phif<<<NODE, 128, 0, stream>>>(E1, Wf, V, Fij);
        // fscale = mlp2(msij)   (second GEMM in-place in E2)
        gemm_wmma<8><<<GE, 128, 0, stream>>>(E1, FF, WT(s+3), FF, bfs0, nullptr, E2, FF, 1, SE);
        gemm_wmma<8><<<GE, 128, 0, stream>>>(E2, FF, WT(s+4), FF, bfs1, nullptr, E2, FF, 0, SE);
        k_Fi<<<NODE, 128, 0, stream>>>(E2, Fij, NM, F_i);
        // phi_r(a)
        gemm_wmma<8><<<GN, 128, 0, stream>>>(a_buf, FF, WT(s+5), FF, br0, nullptr, ntmp, FF, 1, SN);
        gemm_wmma<8><<<GN, 128, 0, stream>>>(ntmp,  FF, WT(s+6), FF, br1, nullptr, pr,   FF, 0, SN);
        // phi_r_ext(msij), no bias
        gemm_wmma<8><<<GE, 128, 0, stream>>>(E1, FF, WT(s+7), FF, nullptr, nullptr, E2, FF, 1, SE);
        gemm_wmma<8><<<GE, 128, 0, stream>>>(E2, FF, WT(s+8), FF, nullptr, nullptr, E2, FF, 0, SE);
        k_drext<<<NODE, 128, 0, stream>>>(E2, Nbr, NM, r_dyn, drext);
        k_update_dyn<<<NODE, 128, 0, stream>>>(F_i, drext, pr, f_dyn, r_dyn);
        // phi_e(a), energy update
        gemm_wmma<8><<<GN, 128, 0, stream>>>(a_buf, FF, WT(s+9), FF, be0, nullptr, ntmp, FF, 1, SN);
        gemm_wmma<8><<<GN, 128, 0, stream>>>(ntmp,  FF, WT(s+10), FF, be1, nullptr, pe,  FF, 0, SN);
        k_energy<<<NODE, 128, 0, stream>>>(pe, f_dyn, r_dyn, a_buf);
    }

    // readout head
    gemm_wmma<8><<<GN, 128, 0, stream>>>(a_buf, FF, WT(33), FF, bae0, nullptr, ntmp, FF, 1, SN);
    gemm_wmma<4><<<GN, 128, 0, stream>>>(ntmp,  FF, WT(34), FF, bae1, nullptr, h2,   64, 1, SN);
    k_final<<<BB, 64, 0, stream>>>(h2, Wae2, bae2, AM, (float*)d_out);
}
